// GatedDeltaNet_30288109371473
// MI455X (gfx1250) — matrix operand order repaired and checked by reading, not compiled, not run on hardware
//
#include <hip/hip_runtime.h>
#include <hip/hip_bf16.h>

typedef __attribute__((ext_vector_type(16))) __bf16 v16bf;
typedef __attribute__((ext_vector_type(8)))  __bf16 v8bf;
typedef __attribute__((ext_vector_type(8)))  float  v8f;
typedef __attribute__((ext_vector_type(4)))  int    v4i;

#define D_DIM 1024
#define T_DIM 2048
#define MT_DIM 16384   // B*T
#define EPSN  1e-8f
#define LDS_S 40       // padded K-stride in LDS (bf16 elems; 80B, 16B-aligned)
#define BK    32

// ---- async global->LDS path (guarded; falls back to reg staging) ----------
#if defined(__has_builtin)
#  if __has_builtin(__builtin_amdgcn_global_load_async_to_lds_b128) && \
      __has_builtin(__builtin_amdgcn_s_wait_asynccnt)
#    define USE_ASYNC 1
#  endif
#endif
#ifndef USE_ASYNC
#  define USE_ASYNC 0
#endif

#if USE_ASYNC
typedef __attribute__((address_space(1))) v4i g_v4i;   // global int4
typedef __attribute__((address_space(3))) v4i l_v4i;   // LDS int4
#endif

__device__ __forceinline__ float sigm(float x) { return 1.0f / (1.0f + __expf(-x)); }

__device__ __forceinline__ __bf16 to_bf16(float f) {
    unsigned u = __builtin_bit_cast(unsigned, f);
    u += 0x7fffu + ((u >> 16) & 1u);                  // round-to-nearest-even
    unsigned short s = (unsigned short)(u >> 16);
    return __builtin_bit_cast(__bf16, s);
}

// 256-thread (8-wave, wave32) block sum reduction
__device__ __forceinline__ float block_reduce_sum(float v, float* sred) {
    #pragma unroll
    for (int off = 16; off > 0; off >>= 1) v += __shfl_down(v, off, 32);
    int lane = threadIdx.x & 31, wid = threadIdx.x >> 5;
    if (lane == 0) sred[wid] = v;
    __syncthreads();
    float s = (threadIdx.x < 8) ? sred[threadIdx.x] : 0.0f;
    if (wid == 0) {
        #pragma unroll
        for (int off = 4; off > 0; off >>= 1) s += __shfl_down(s, off, 32);
        if (lane == 0) sred[0] = s;
    }
    __syncthreads();
    float r = sred[0];
    __syncthreads();  // allow sred reuse
    return r;
}

// --------------------------------------------------------------------------
// Kernel 1: h = zc_rms(x, g1) -> bf16 ; also x -> bf16
// --------------------------------------------------------------------------
__global__ __launch_bounds__(256) void prep_norm(const float* __restrict__ x,
                                                 const float* __restrict__ g1,
                                                 __bf16* __restrict__ hb,
                                                 __bf16* __restrict__ xb) {
    __shared__ float sred[8];
    size_t row = blockIdx.x;
    const float* xr = x + row * D_DIM;
    float v[4]; float s = 0.0f;
    #pragma unroll
    for (int i = 0; i < 4; ++i) { int d = threadIdx.x + i * 256; v[i] = xr[d]; s += v[i]; }
    float mean = block_reduce_sum(s, sred) * (1.0f / D_DIM);
    float ss = 0.0f;
    #pragma unroll
    for (int i = 0; i < 4; ++i) { float c = v[i] - mean; ss += c * c; }
    float rstd = rsqrtf(block_reduce_sum(ss, sred) * (1.0f / D_DIM) + EPSN);
    #pragma unroll
    for (int i = 0; i < 4; ++i) {
        int d = threadIdx.x + i * 256;
        hb[row * D_DIM + d] = to_bf16((v[i] - mean) * rstd * g1[d]);
        xb[row * D_DIM + d] = to_bf16(v[i]);
    }
}

// --------------------------------------------------------------------------
// Kernel 2: fp32 -> bf16 weight conversion WITH transpose:
// dst[n][k] = bf16(src[k][n]) so GEMM B-tiles stage as contiguous rows.
// --------------------------------------------------------------------------
__global__ __launch_bounds__(256) void cvt_bf16_t(const float* __restrict__ src,
                                                  __bf16* __restrict__ dst) {
    int i = blockIdx.x * 256 + threadIdx.x;        // over D*D
    int k = i >> 10, n = i & (D_DIM - 1);
    dst[(size_t)n * D_DIM + k] = to_bf16(src[i]);
}

// --------------------------------------------------------------------------
// Stage a 128-row x 32-col bf16 tile (row-major, ld = D_DIM) into LDS
// (row stride LDS_S). Async to LDS if available, else reg round-trip.
// --------------------------------------------------------------------------
__device__ __forceinline__ void stage_tile(__bf16* __restrict__ lds,
                                           const __bf16* __restrict__ g, int tid) {
    #pragma unroll
    for (int it = 0; it < 2; ++it) {
        int idx = (it * 256 + tid) * 8;            // 8 bf16 (16B) per thread per it
        int r = idx >> 5, cc = idx & 31;
        const __bf16* gp = g + (size_t)r * D_DIM + cc;
        __bf16* lp = lds + r * LDS_S + cc;
#if USE_ASYNC
        __builtin_amdgcn_global_load_async_to_lds_b128(
            (g_v4i*)(unsigned long long)gp,
            (l_v4i*)(unsigned)(unsigned long long)lp, 0, 0);
#else
        *(uint4*)lp = *(const uint4*)gp;
#endif
    }
}

__device__ __forceinline__ void wait_stage() {
#if USE_ASYNC
    __builtin_amdgcn_s_wait_asynccnt(0);
#endif
    __syncthreads();
}

// --------------------------------------------------------------------------
// Kernel 3: C[M,N] = A[M,K] @ W[K,N] + bias  (M=16384, N=K=1024)
// A row-major bf16, Bt = W transposed ([n][k]) bf16. f32 WMMA accumulate.
// Workgroup tile 128x128, K-step 32, double-buffered LDS, 1 barrier/tile.
// 8 waves: wave (wm 0..3, wn 0..1) owns 32x64 = 2x4 WMMA tiles.
// EPI==1: fused out = x + sigmoid(silu(dhat))*dhat
// --------------------------------------------------------------------------
template <int EPI>
__global__ __launch_bounds__(256) void gemm_wmma(const __bf16* __restrict__ A,
                                                 const __bf16* __restrict__ Bt,
                                                 const float* __restrict__ bias,
                                                 float* __restrict__ Cout,
                                                 const float* __restrict__ Xres,
                                                 float* __restrict__ Out) {
    __shared__ __bf16 As[2][128][LDS_S];   // [buf][m][k]
    __shared__ __bf16 Bs[2][128][LDS_S];   // [buf][n][k]

    const int tid  = threadIdx.x;
    const int wave = tid >> 5, lane = tid & 31;
    const int wm = wave & 3, wn = wave >> 2;
    const int m0 = blockIdx.y * 128, n0 = blockIdx.x * 128;

    v8f c[2][4] = {};

    const int rowA  = wm * 32 + (lane & 15);
    const int koffA = (lane >> 4) * 8;
    const int colB  = wn * 64 + (lane & 15);
    const int koffB = (lane >> 4) * 8;

    const __bf16* Ag = A  + (size_t)m0 * D_DIM;
    const __bf16* Bg = Bt + (size_t)n0 * D_DIM;

    stage_tile(&As[0][0][0], Ag, tid);
    stage_tile(&Bs[0][0][0], Bg, tid);

    int buf = 0;
    for (int k0 = 0; k0 < D_DIM; k0 += BK) {
        wait_stage();                       // staged tile complete on all waves
        if (k0 + BK < D_DIM) {              // prefetch next tile into other buf
            stage_tile(&As[buf ^ 1][0][0], Ag + k0 + BK, tid);
            stage_tile(&Bs[buf ^ 1][0][0], Bg + k0 + BK, tid);
        }

        v16bf a[2], b[4];
        #pragma unroll
        for (int i = 0; i < 2; ++i) {
            const __bf16* p = &As[buf][rowA + i * 16][koffA];
            v8bf lo = *(const v8bf*)p;
            v8bf hi = *(const v8bf*)(p + 16);
            #pragma unroll
            for (int e = 0; e < 8; ++e) { a[i][e] = lo[e]; a[i][e + 8] = hi[e]; }
        }
        #pragma unroll
        for (int j = 0; j < 4; ++j) {
            const __bf16* p = &Bs[buf][colB + j * 16][koffB];
            v8bf lo = *(const v8bf*)p;
            v8bf hi = *(const v8bf*)(p + 16);
            #pragma unroll
            for (int e = 0; e < 8; ++e) { b[j][e] = lo[e]; b[j][e + 8] = hi[e]; }
        }
        #pragma unroll
        for (int i = 0; i < 2; ++i)
            #pragma unroll
            for (int j = 0; j < 4; ++j)
                c[i][j] = __builtin_amdgcn_wmma_f32_16x16x32_bf16(
                    false, a[i], false, b[j], (short)0, c[i][j], false, false);

        buf ^= 1;
    }

    // ---- epilogue ----
    const int rbase = m0 + wm * 32 + (lane >> 4) * 8;
    const int cbase = n0 + wn * 64 + (lane & 15);
    #pragma unroll
    for (int i = 0; i < 2; ++i) {
        #pragma unroll
        for (int j = 0; j < 4; ++j) {
            int col = cbase + j * 16;
            float bia = bias[col];
            #pragma unroll
            for (int e = 0; e < 8; ++e) {
                size_t row = rbase + i * 16 + e;
                float val = c[i][j][e] + bia;
                if (EPI == 0) {
                    Cout[row * D_DIM + col] = val;
                } else {
                    float sil  = val * sigm(val);   // silu(dhat)
                    float gate = sigm(sil);         // sigmoid(silu(dhat))
                    Out[row * D_DIM + col] = Xres[row * D_DIM + col] + gate * val;
                }
            }
        }
    }
}

// --------------------------------------------------------------------------
// Kernel 4: dwconv3+sigmoid on q/k/v, l2norm(q,k), delta, zc_rms -> bf16 dn
// --------------------------------------------------------------------------
__global__ __launch_bounds__(256) void mix(const float* __restrict__ qpre,
                                           const float* __restrict__ kpre,
                                           const float* __restrict__ vpre,
                                           const float* __restrict__ apre,
                                           const float* __restrict__ bpre,
                                           const float* __restrict__ wqc, const float* __restrict__ bqc,
                                           const float* __restrict__ wkc, const float* __restrict__ bkc,
                                           const float* __restrict__ wvc, const float* __restrict__ bvc,
                                           const float* __restrict__ g2,
                                           __bf16* __restrict__ dn) {
    __shared__ float sred[8];
    size_t row = blockIdx.x;
    int t = (int)(row % T_DIM);
    bool hp = (t > 0), hn = (t < T_DIM - 1);
    size_t rp = row * D_DIM, rm = rp - D_DIM, rn = rp + D_DIM;

    float q[4], k[4], v[4], dl[4];
    float sq = 0.0f, sk = 0.0f;
    #pragma unroll
    for (int i = 0; i < 4; ++i) {
        int d = threadIdx.x + i * 256;
        float qm = hp ? qpre[rm + d] : 0.0f, qc = qpre[rp + d], qn = hn ? qpre[rn + d] : 0.0f;
        float km = hp ? kpre[rm + d] : 0.0f, kc = kpre[rp + d], kn = hn ? kpre[rn + d] : 0.0f;
        float vm = hp ? vpre[rm + d] : 0.0f, vc = vpre[rp + d], vn = hn ? vpre[rn + d] : 0.0f;
        q[i] = sigm(wqc[d*3+0]*qm + wqc[d*3+1]*qc + wqc[d*3+2]*qn + bqc[d]);
        k[i] = sigm(wkc[d*3+0]*km + wkc[d*3+1]*kc + wkc[d*3+2]*kn + bkc[d]);
        v[i] = sigm(wvc[d*3+0]*vm + wvc[d*3+1]*vc + wvc[d*3+2]*vn + bvc[d]);
        sq += q[i] * q[i];
        sk += k[i] * k[i];
    }
    float rq = rsqrtf(block_reduce_sum(sq, sred) + EPSN);
    float rk = rsqrtf(block_reduce_sum(sk, sred) + EPSN);

    float sm = 0.0f;
    #pragma unroll
    for (int i = 0; i < 4; ++i) {
        int d = threadIdx.x + i * 256;
        float delta = (q[i] * rq) * ((k[i] * rk) * v[i]);
        dl[i] = tanhf(apre[rp + d]) * delta + bpre[rp + d];
        sm += dl[i];
    }
    float mean = block_reduce_sum(sm, sred) * (1.0f / D_DIM);
    float ss = 0.0f;
    #pragma unroll
    for (int i = 0; i < 4; ++i) { float c = dl[i] - mean; ss += c * c; }
    float rstd = rsqrtf(block_reduce_sum(ss, sred) * (1.0f / D_DIM) + EPSN);
    #pragma unroll
    for (int i = 0; i < 4; ++i) {
        int d = threadIdx.x + i * 256;
        dn[rp + d] = to_bf16((dl[i] - mean) * rstd * g2[d]);
    }
}

// --------------------------------------------------------------------------
extern "C" void kernel_launch(void* const* d_in, const int* in_sizes, int n_in,
                              void* d_out, int out_size, void* d_ws, size_t ws_size,
                              hipStream_t stream) {
    const float* x   = (const float*)d_in[0];
    const float* g1  = (const float*)d_in[1];
    const float* Wq  = (const float*)d_in[2];
    const float* bq  = (const float*)d_in[3];
    const float* Wk  = (const float*)d_in[4];
    const float* bk  = (const float*)d_in[5];
    const float* Wv  = (const float*)d_in[6];
    const float* bv  = (const float*)d_in[7];
    const float* wqc = (const float*)d_in[8];
    const float* bqc = (const float*)d_in[9];
    const float* wkc = (const float*)d_in[10];
    const float* bkc = (const float*)d_in[11];
    const float* wvc = (const float*)d_in[12];
    const float* bvc = (const float*)d_in[13];
    const float* Wa  = (const float*)d_in[14];
    const float* ba  = (const float*)d_in[15];
    const float* Wb  = (const float*)d_in[16];
    const float* bb  = (const float*)d_in[17];
    const float* g2  = (const float*)d_in[18];
    const float* Wp  = (const float*)d_in[19];
    const float* bp  = (const float*)d_in[20];

    char* ws = (char*)d_ws;
    const size_t ACT_B = (size_t)MT_DIM * D_DIM * sizeof(__bf16);   // 32 MB
    const size_t WB    = (size_t)D_DIM * D_DIM * sizeof(__bf16);    //  2 MB
    const size_t ACT_F = (size_t)MT_DIM * D_DIM * sizeof(float);    // 64 MB

    __bf16* hb   = (__bf16*)(ws);                 // also reused for dn later
    __bf16* xb   = (__bf16*)(ws + ACT_B);
    __bf16* wqb  = (__bf16*)(ws + 2*ACT_B);
    __bf16* wkb  = (__bf16*)(ws + 2*ACT_B + 1*WB);
    __bf16* wvb  = (__bf16*)(ws + 2*ACT_B + 2*WB);
    __bf16* wab  = (__bf16*)(ws + 2*ACT_B + 3*WB);
    __bf16* wbb  = (__bf16*)(ws + 2*ACT_B + 4*WB);
    __bf16* wpb  = (__bf16*)(ws + 2*ACT_B + 5*WB);
    char*   pre0 = ws + 2*ACT_B + 6*WB;
    float* qpre = (float*)(pre0);
    float* kpre = (float*)(pre0 + 1*ACT_F);
    float* vpre = (float*)(pre0 + 2*ACT_F);
    float* apre = (float*)(pre0 + 3*ACT_F);
    float* bpre = (float*)(pre0 + 4*ACT_F);

    // 1. norm + bf16 conversions (weights converted transposed)
    prep_norm<<<MT_DIM, 256, 0, stream>>>(x, g1, hb, xb);
    const int NWBLK = (D_DIM * D_DIM) / 256;
    cvt_bf16_t<<<NWBLK, 256, 0, stream>>>(Wq, wqb);
    cvt_bf16_t<<<NWBLK, 256, 0, stream>>>(Wk, wkb);
    cvt_bf16_t<<<NWBLK, 256, 0, stream>>>(Wv, wvb);
    cvt_bf16_t<<<NWBLK, 256, 0, stream>>>(Wa, wab);
    cvt_bf16_t<<<NWBLK, 256, 0, stream>>>(Wb, wbb);
    cvt_bf16_t<<<NWBLK, 256, 0, stream>>>(Wp, wpb);

    // 2. five projection GEMMs (WMMA bf16 -> f32)
    dim3 ggrid(D_DIM / 128, MT_DIM / 128);  // (8, 128)
    gemm_wmma<0><<<ggrid, 256, 0, stream>>>(hb, wqb, bq, qpre, nullptr, nullptr);
    gemm_wmma<0><<<ggrid, 256, 0, stream>>>(hb, wkb, bk, kpre, nullptr, nullptr);
    gemm_wmma<0><<<ggrid, 256, 0, stream>>>(hb, wvb, bv, vpre, nullptr, nullptr);
    gemm_wmma<0><<<ggrid, 256, 0, stream>>>(xb, wab, ba, apre, nullptr, nullptr);
    gemm_wmma<0><<<ggrid, 256, 0, stream>>>(xb, wbb, bb, bpre, nullptr, nullptr);

    // 3. conv/gate/norm fusion -> dn (bf16, reuses hb)
    mix<<<MT_DIM, 256, 0, stream>>>(qpre, kpre, vpre, apre, bpre,
                                    wqc, bqc, wkc, bkc, wvc, bvc, g2, hb);

    // 4. final GEMM with fused residual gate epilogue
    gemm_wmma<1><<<ggrid, 256, 0, stream>>>(hb, wpb, bp, nullptr, x, (float*)d_out);
}
